// SingleDomainHead_56057913147555
// MI455X (gfx1250) — matrix operand order repaired and checked by reading, not compiled, not run on hardware
//
#include <hip/hip_runtime.h>
#include <math.h>

// ---------- types ----------
typedef __bf16 bf16_t;
typedef __attribute__((ext_vector_type(4)))  __bf16 v4bf;
typedef __attribute__((ext_vector_type(8)))  __bf16 v8bf;
typedef __attribute__((ext_vector_type(16))) __bf16 v16bf;
typedef __attribute__((ext_vector_type(8)))  float  v8f;

#define HDIM 2048
#define BDIM 8192
#define KPROTO 64
#define ADIM 64

static __device__ __forceinline__ v16bf cat16(v8bf lo, v8bf hi) {
  return __builtin_shufflevector(lo, hi, 0,1,2,3,4,5,6,7,8,9,10,11,12,13,14,15);
}

// low 32 bits of a generic shared pointer == LDS byte offset (aperture is in [63:32])
static __device__ __forceinline__ unsigned lds_off(const void* p) {
  return (unsigned)(size_t)p;
}

// ---------- prep: h_bf16 = bf16(h) (also generic f32 -> bf16 vector convert) ----------
__global__ __launch_bounds__(256) void cvt_bf16_kernel(
    const float* __restrict__ src, bf16_t* __restrict__ dst, int n4) {
  int i = blockIdx.x * 256 + threadIdx.x;
  if (i < n4) {
    float4 a = ((const float4*)src)[i];
    v4bf o;
    o[0] = (bf16_t)a.x; o[1] = (bf16_t)a.y;
    o[2] = (bf16_t)a.z; o[3] = (bf16_t)a.w;
    ((v4bf*)dst)[i] = o;
  }
}

// ---------- prep: W1b[k*H+n] = bf16(proj_W[n*H+k])  (B-layout for GEMM1) ----------
__global__ __launch_bounds__(256) void transpose_bf16_kernel(
    const float* __restrict__ src, bf16_t* __restrict__ dst, int H) {
  __shared__ float tile[64][65];
  const int bn = blockIdx.y * 64;  // source row block (n)
  const int bk = blockIdx.x * 64;  // source col block (k)
  const int r  = threadIdx.x >> 2;         // 0..63
  const int c0 = (threadIdx.x & 3) * 16;   // 0,16,32,48
  const float4* s = (const float4*)(src + (size_t)(bn + r) * H + bk + c0);
  #pragma unroll
  for (int q = 0; q < 4; ++q) {
    float4 v = s[q];
    tile[r][c0 + q*4 + 0] = v.x;
    tile[r][c0 + q*4 + 1] = v.y;
    tile[r][c0 + q*4 + 2] = v.z;
    tile[r][c0 + q*4 + 3] = v.w;
  }
  __syncthreads();
  v8bf o0, o1;
  #pragma unroll
  for (int i = 0; i < 8; ++i) {
    o0[i] = (bf16_t)tile[c0 + i][r];
    o1[i] = (bf16_t)tile[c0 + 8 + i][r];
  }
  v8bf* d = (v8bf*)(dst + (size_t)(bk + r) * H + bn + c0);
  d[0] = o0; d[1] = o1;
}

// ---------- prep: W2b = bf16(syn_W * mask); also zero entropy accumulator ----------
__global__ __launch_bounds__(256) void prep_w2_kernel(
    const float* __restrict__ syn, const float* __restrict__ mask,
    bf16_t* __restrict__ W2b, float* __restrict__ ent_out, int n4) {
  int i = blockIdx.x * 256 + threadIdx.x;
  if (i == 0) *ent_out = 0.0f;
  if (i < n4) {
    float4 a = ((const float4*)syn)[i];
    float4 m = ((const float4*)mask)[i];
    v4bf o;
    o[0] = (bf16_t)(a.x * m.x);
    o[1] = (bf16_t)(a.y * m.y);
    o[2] = (bf16_t)(a.z * m.z);
    o[3] = (bf16_t)(a.w * m.w);
    ((v4bf*)W2b)[i] = o;
  }
}

// ---------- prep: Pb[k*K + j] = bf16(l2norm(protos)[j][k]) ----------
__global__ __launch_bounds__(256) void proto_kernel(
    const float* __restrict__ protos, bf16_t* __restrict__ Pb, int H, int Kp) {
  __shared__ float red[8];
  const int j = blockIdx.x;
  const float* row = protos + (size_t)j * H;
  float ssq = 0.0f;
  for (int k = threadIdx.x; k < H; k += 256) { float x = row[k]; ssq += x * x; }
  #pragma unroll
  for (int off = 16; off; off >>= 1) ssq += __shfl_xor(ssq, off);
  if ((threadIdx.x & 31) == 0) red[threadIdx.x >> 5] = ssq;
  __syncthreads();
  float tot = 0.0f;
  #pragma unroll
  for (int q = 0; q < 8; ++q) tot += red[q];
  const float inv = 1.0f / fmaxf(sqrtf(tot), 1e-12f);
  for (int k = threadIdx.x; k < H; k += 256)
    Pb[(size_t)k * Kp + j] = (bf16_t)(row[k] * inv);
}

// ---------- main GEMM: C(bf16) = A(bf16) x B(bf16), 128x128x32 tiles ----------
// Double-buffered LDS staging via GLOBAL_LOAD_ASYNC_TO_LDS_B128 (ASYNCcnt),
// compute on v_wmma_f32_16x16x32_bf16.
__global__ __launch_bounds__(256) void gemm_async_kernel(
    const bf16_t* __restrict__ A, const bf16_t* __restrict__ B,
    bf16_t* __restrict__ C, int M, int N, int K) {
  (void)M;
  __shared__ __align__(16) bf16_t As[2][128][40];   // pad 8 -> conflict-free frag reads
  __shared__ __align__(16) bf16_t Bs[2][32][136];
  const int tid  = threadIdx.x;
  const int lane = tid & 31;
  const int w    = tid >> 5;
  const int wm   = w & 3;       // wave rows: wm*32 .. +31
  const int wn   = w >> 2;      // wave cols: wn*64 .. +63
  const int bm   = blockIdx.y * 128;
  const int bn   = blockIdx.x * 128;

  v8f acc[2][4] = {};

  // staging coords: each thread DMAs 32B of A and 32B of B per tile
  const unsigned arow = tid >> 1;              // 0..127
  const unsigned acol = (tid & 1) * 16;        // 0 / 16
  const unsigned brow = tid >> 3;              // 0..31
  const unsigned bcol = (tid & 7) * 16;        // 0..112

  const bf16_t* Asad = A + (size_t)bm * K;     // uniform 64-bit SGPR base
  const bf16_t* Bsad = B + bn;
  const unsigned voffA0 = (arow * (unsigned)K + acol) * 2u;   // per-lane byte offset
  const unsigned voffB0 = (brow * (unsigned)N + bcol) * 2u;

  // issue one tile's async DMA (4 x B128 per thread; offset:16 advances LDS+global)
  auto issue = [&](int buf, int k0) {
    unsigned lA = lds_off(&As[buf][arow][acol]);
    unsigned lB = lds_off(&Bs[buf][brow][bcol]);
    unsigned oA = voffA0 + (unsigned)k0 * 2u;
    unsigned oB = voffB0 + (unsigned)k0 * (unsigned)N * 2u;
    asm volatile(
        "global_load_async_to_lds_b128 %0, %2, %4\n\t"
        "global_load_async_to_lds_b128 %0, %2, %4 offset:16\n\t"
        "global_load_async_to_lds_b128 %1, %3, %5\n\t"
        "global_load_async_to_lds_b128 %1, %3, %5 offset:16"
        :: "v"(lA), "v"(lB), "v"(oA), "v"(oB), "s"(Asad), "s"(Bsad)
        : "memory");
  };

  issue(0, 0);
  int buf = 0;
  for (int k0 = 0; k0 < K; k0 += 32) {
    if (k0 + 32 < K) {
      issue(buf ^ 1, k0 + 32);
      // async loads complete in order: <=4 outstanding => current tile landed
      asm volatile("s_wait_asynccnt 4" ::: "memory");
    } else {
      asm volatile("s_wait_asynccnt 0" ::: "memory");
    }
    __syncthreads();

    // ---- fragments per CDNA5 16-bit WMMA VGPR layouts ----
    const int khalf = (lane >> 4) * 8;
    const int ar0   = wm * 32 + (lane & 15);
    v16bf afrag[2];
    #pragma unroll
    for (int mt = 0; mt < 2; ++mt) {
      v8bf c0 = *(const v8bf*)&As[buf][ar0 + mt * 16][khalf];
      v8bf c1 = *(const v8bf*)&As[buf][ar0 + mt * 16][khalf + 16];
      afrag[mt] = cat16(c0, c1);
    }
    const int ncol = lane & 15;   // B: lane = N column, elements over K
    #pragma unroll
    for (int nt = 0; nt < 4; ++nt) {
      const int cb = wn * 64 + nt * 16;
      v16bf bfrag;
      #pragma unroll
      for (int s = 0; s < 16; ++s)
        bfrag[s] = Bs[buf][khalf + (s & 7) + (s >> 3) * 16][cb + ncol];
      acc[0][nt] = __builtin_amdgcn_wmma_f32_16x16x32_bf16(
          false, afrag[0], false, bfrag, (short)0, acc[0][nt], false, false);
      acc[1][nt] = __builtin_amdgcn_wmma_f32_16x16x32_bf16(
          false, afrag[1], false, bfrag, (short)0, acc[1][nt], false, false);
    }
    __syncthreads();
    buf ^= 1;
  }

  // ---- store C as bf16 (C layout: VGPR v -> M=v (lanes 0-15) / v+8 (lanes 16-31)) ----
  const int rlo = (lane >> 4) * 8;
  #pragma unroll
  for (int mt = 0; mt < 2; ++mt)
    #pragma unroll
    for (int nt = 0; nt < 4; ++nt) {
      const int col = bn + wn * 64 + nt * 16 + (lane & 15);
      #pragma unroll
      for (int v = 0; v < 8; ++v) {
        const int row = bm + wm * 32 + mt * 16 + rlo + v;
        C[(size_t)row * N + col] = (bf16_t)acc[mt][nt][v];
      }
    }
}

// ---------- GEMM3 + row L2 norm: sim_un = h_syn @ Pb ; rnorm = 1/max(||row||,eps) ----------
__global__ __launch_bounds__(256) void gemm3_norm_kernel(
    const bf16_t* __restrict__ A, const bf16_t* __restrict__ Pb,
    float* __restrict__ sim, float* __restrict__ rnorm, int K) {
  __shared__ __align__(16) bf16_t As[128][40];
  __shared__ __align__(16) bf16_t Bs[32][72];
  const int tid  = threadIdx.x;
  const int lane = tid & 31;
  const int w    = tid >> 5;      // wave owns rows w*16 .. w*16+15
  const int bm   = blockIdx.x * 128;

  v8f acc[4] = {};
  float ssq = 0.0f;

  const int arow = tid >> 1;
  const int acol = (tid & 1) * 16;
  const int brow = tid >> 3;          // 0..31
  const int bcol = (tid & 7) * 8;     // 0..56

  for (int k0 = 0; k0 < K; k0 += 32) {
    const v8bf* as = (const v8bf*)(A + (size_t)(bm + arow) * K + k0 + acol);
    v8bf a0 = as[0], a1 = as[1];
    *(v8bf*)&As[arow][acol]     = a0;
    *(v8bf*)&As[arow][acol + 8] = a1;
    #pragma unroll
    for (int i = 0; i < 8; ++i) {
      float x = (float)a0[i]; ssq += x * x;
      float y = (float)a1[i]; ssq += y * y;
    }
    const v8bf* bs = (const v8bf*)(Pb + (size_t)(k0 + brow) * KPROTO + bcol);
    *(v8bf*)&Bs[brow][bcol] = bs[0];
    __syncthreads();

    const int khalf = (lane >> 4) * 8;
    const int ar0   = w * 16 + (lane & 15);
    v8bf c0 = *(const v8bf*)&As[ar0][khalf];
    v8bf c1 = *(const v8bf*)&As[ar0][khalf + 16];
    v16bf afrag = cat16(c0, c1);
    const int ncol = lane & 15;
    #pragma unroll
    for (int nt = 0; nt < 4; ++nt) {
      v16bf bfrag;
      #pragma unroll
      for (int s = 0; s < 16; ++s)
        bfrag[s] = Bs[khalf + (s & 7) + (s >> 3) * 16][nt * 16 + ncol];
      acc[nt] = __builtin_amdgcn_wmma_f32_16x16x32_bf16(
          false, afrag, false, bfrag, (short)0, acc[nt], false, false);
    }
    __syncthreads();
  }

  // row norm: partner thread (tid^1) holds the other half of the same row
  ssq += __shfl_xor(ssq, 1);
  if ((tid & 1) == 0)
    rnorm[bm + (tid >> 1)] = 1.0f / fmaxf(sqrtf(ssq), 1e-12f);

  const int rlo = (lane >> 4) * 8;
  #pragma unroll
  for (int nt = 0; nt < 4; ++nt)
    #pragma unroll
    for (int v = 0; v < 8; ++v) {
      const int row = bm + w * 16 + rlo + v;
      sim[(size_t)row * KPROTO + nt * 16 + (lane & 15)] = acc[nt][v];
    }
}

// ---------- softmax + entropy + argmax + action matvec (one wave per row) ----------
__global__ __launch_bounds__(256) void softmax_action_kernel(
    const float* __restrict__ sim, const float* __restrict__ rnorm,
    const float* __restrict__ gates, const float* __restrict__ actions,
    float* __restrict__ action_out, float* __restrict__ ent_out,
    float* __restrict__ top_out, int B) {
  __shared__ float wsm[8][64];
  const int w = threadIdx.x >> 5, lane = threadIdx.x & 31;
  const int b = blockIdx.x * 8 + w;
  const float rn = rnorm[b];
  float s0 = sim[(size_t)b * 64 + lane]      * rn * gates[lane];
  float s1 = sim[(size_t)b * 64 + lane + 32] * rn * gates[lane + 32];

  float mv; int mi;
  if (s0 >= s1) { mv = s0; mi = lane; } else { mv = s1; mi = lane + 32; }
  #pragma unroll
  for (int off = 16; off; off >>= 1) {
    float ov = __shfl_xor(mv, off);
    int   oi = __shfl_xor(mi, off);
    if (ov > mv || (ov == mv && oi < mi)) { mv = ov; mi = oi; }
  }
  const float e0 = __expf(s0 - mv), e1 = __expf(s1 - mv);
  float sum = e0 + e1;
  #pragma unroll
  for (int off = 16; off; off >>= 1) sum += __shfl_xor(sum, off);
  const float inv = 1.0f / sum;
  const float w0 = e0 * inv, w1 = e1 * inv;
  wsm[w][lane] = w0; wsm[w][lane + 32] = w1;

  float ent = -(w0 * __logf(w0 + 1e-8f) + w1 * __logf(w1 + 1e-8f));
  #pragma unroll
  for (int off = 16; off; off >>= 1) ent += __shfl_xor(ent, off);
  if (lane == 0) {
    atomicAdd(ent_out, ent / (float)B);
    top_out[b] = (float)mi;
  }

  float a0 = 0.0f, a1 = 0.0f;
  #pragma unroll
  for (int j = 0; j < 64; ++j) {
    const float wj = wsm[w][j];
    a0 += wj * actions[j * 64 + lane];
    a1 += wj * actions[j * 64 + lane + 32];
  }
  action_out[(size_t)b * 64 + lane]      = a0;
  action_out[(size_t)b * 64 + lane + 32] = a1;
}

// ---------- launch ----------
extern "C" void kernel_launch(void* const* d_in, const int* in_sizes, int n_in,
                              void* d_out, int out_size, void* d_ws, size_t ws_size,
                              hipStream_t stream) {
  (void)in_sizes; (void)n_in; (void)out_size; (void)ws_size;
  const float* h       = (const float*)d_in[0];
  const float* proj_W  = (const float*)d_in[1];
  const float* syn_W   = (const float*)d_in[2];
  const float* mask    = (const float*)d_in[3];
  const float* protos  = (const float*)d_in[4];
  const float* actions = (const float*)d_in[5];
  const float* gates   = (const float*)d_in[6];

  float* out        = (float*)d_out;
  float* action_out = out;                         // 8192*64
  float* ent_out    = out + (size_t)BDIM * ADIM;   // 1
  float* top_out    = ent_out + 1;                 // 8192

  char* ws = (char*)d_ws;
  auto alloc = [&](size_t bytes) -> char* {
    char* p = ws;
    ws += (bytes + 255) & ~(size_t)255;
    return p;
  };
  bf16_t* W1b   = (bf16_t*)alloc((size_t)HDIM * HDIM * 2);
  bf16_t* W2b   = (bf16_t*)alloc((size_t)HDIM * HDIM * 2);
  bf16_t* Pb    = (bf16_t*)alloc((size_t)HDIM * KPROTO * 2);
  bf16_t* h_b   = (bf16_t*)alloc((size_t)BDIM * HDIM * 2);
  bf16_t* h_in  = (bf16_t*)alloc((size_t)BDIM * HDIM * 2);
  bf16_t* h_syn = (bf16_t*)alloc((size_t)BDIM * HDIM * 2);
  float*  sim   = (float*) alloc((size_t)BDIM * KPROTO * 4);
  float*  rnorm = (float*) alloc((size_t)BDIM * 4);

  // prep
  cvt_bf16_kernel<<<(BDIM * HDIM / 4 + 255) / 256, 256, 0, stream>>>(
      h, h_b, BDIM * HDIM / 4);
  transpose_bf16_kernel<<<dim3(HDIM / 64, HDIM / 64), 256, 0, stream>>>(proj_W, W1b, HDIM);
  prep_w2_kernel<<<(HDIM * HDIM / 4 + 255) / 256, 256, 0, stream>>>(
      syn_W, mask, W2b, ent_out, HDIM * HDIM / 4);
  proto_kernel<<<KPROTO, 256, 0, stream>>>(protos, Pb, HDIM, KPROTO);

  // GEMM1: h_in = h @ proj_W^T
  gemm_async_kernel<<<dim3(HDIM / 128, BDIM / 128), 256, 0, stream>>>(
      h_b, W1b, h_in, BDIM, HDIM, HDIM);
  // GEMM2: h_syn = h_in @ (syn_W*mask)
  gemm_async_kernel<<<dim3(HDIM / 128, BDIM / 128), 256, 0, stream>>>(
      h_in, W2b, h_syn, BDIM, HDIM, HDIM);
  // GEMM3 + row norms
  gemm3_norm_kernel<<<BDIM / 128, 256, 0, stream>>>(h_syn, Pb, sim, rnorm, HDIM);
  // softmax / entropy / argmax / action
  softmax_action_kernel<<<BDIM / 8, 256, 0, stream>>>(
      sim, rnorm, gates, actions, action_out, ent_out, top_out, BDIM);
}
